// AttentionAT_75514114998391
// MI455X (gfx1250) — matrix operand order repaired and checked by reading, not compiled, not run on hardware
//
#include <hip/hip_runtime.h>

typedef __bf16 bf16_t;
typedef __attribute__((ext_vector_type(16))) __bf16 v16bf;
typedef __attribute__((ext_vector_type(8)))  __bf16 v8bf;
typedef __attribute__((ext_vector_type(4)))  __bf16 v4bf;
typedef __attribute__((ext_vector_type(8)))  float  v8f;
typedef __attribute__((ext_vector_type(4)))  unsigned int u32x4;
typedef __attribute__((ext_vector_type(4)))  int i32x4;
typedef __attribute__((ext_vector_type(8)))  int i32x8;

#define DIMC   768
#define SEQN   4096
#define NBATCH 8
#define SCALEF 0.10206207261596577f   // (768/8)^-0.5

#define TILE   128
#define KT     32
#define LPITCH 40   // bf16 elems per LDS row: 32 data + 8 pad (= 80 B, 16B aligned)

#if defined(__has_builtin)
#  if __has_builtin(__builtin_amdgcn_tensor_load_to_lds) && __has_builtin(__builtin_amdgcn_s_wait_tensorcnt)
#    define HAVE_TDM 1
#  endif
#endif
#ifndef HAVE_TDM
#  define HAVE_TDM 0
#endif

// Build a 16-wide bf16 fragment from two 16-byte LDS runs (2 x ds_load_b128).
__device__ __forceinline__ v16bf load_frag(const bf16_t* row, int off0, int off1) {
    v8bf lo = *(const v8bf*)(row + off0);
    v8bf hi = *(const v8bf*)(row + off1);
    return __builtin_shufflevector(lo, hi, 0,1,2,3,4,5,6,7,8,9,10,11,12,13,14,15);
}

// ---------------------------------------------------------------------------
// Kernel 0: one-shot fp32 -> bf16 conversion of x_2 (V) into workspace so the
// PV GEMM can stage its B tiles with the Tensor Data Mover (no conversion).
// ---------------------------------------------------------------------------
__global__ __launch_bounds__(256)
void cvt_bf16_kernel(const float* __restrict__ X, bf16_t* __restrict__ Y)
{
    size_t i = (size_t)blockIdx.x * 256 + threadIdx.x;       // one float4 each
    float4 v = *(const float4*)(X + i * 4);
    v4bf p = { (bf16_t)v.x, (bf16_t)v.y, (bf16_t)v.z, (bf16_t)v.w };
    *(v4bf*)(Y + i * 4) = p;
}

// ---------------------------------------------------------------------------
// Kernel 1:  S[b,c,d] = sum_n Q[b,n,c] * K[b,n,d]      (raw logits, no scale)
// A = Q^T (M=c, K=n)  -> staged transposed into LDS [c][n]  (TDM can't
// transpose, so this stays on the cvt+ds_store_b16 path)
// ---------------------------------------------------------------------------
__global__ __launch_bounds__(256)
void qk_gemm_kernel(const float* __restrict__ Q, const float* __restrict__ K,
                    float* __restrict__ S)
{
    __shared__ __attribute__((aligned(16))) bf16_t lq[TILE * LPITCH];
    __shared__ __attribute__((aligned(16))) bf16_t lk[TILE * LPITCH];

    const int b    = blockIdx.z;
    const int c0   = blockIdx.x * TILE;
    const int d0   = blockIdx.y * TILE;
    const int t    = threadIdx.x;
    const int wave = t >> 5, lane = t & 31;
    const int wm   = wave & 3, wn = wave >> 2;   // 4x2 waves over 128x128
    const int lr   = lane & 15, lh = lane >> 4;

    const float* Qb = Q + (size_t)b * SEQN * DIMC;
    const float* Kb = K + (size_t)b * SEQN * DIMC;

    v8f acc[2][4] = {};

    for (int n0 = 0; n0 < SEQN; n0 += KT) {
        if (n0 + KT < SEQN) {   // gfx1250 global_prefetch_b8 for next K-step
            __builtin_prefetch(Qb + (size_t)(n0 + KT) * DIMC + c0, 0, 1);
            __builtin_prefetch(Kb + (size_t)(n0 + KT) * DIMC + d0, 0, 1);
        }
        __syncthreads();
        #pragma unroll
        for (int i = 0; i < 4; ++i) {
            int idx = t + i * 256;          // 0..1023
            int r   = idx >> 5;             // n-local 0..31
            int q   = idx & 31;             // group of 4 channels
            float4 qv = *(const float4*)(Qb + (size_t)(n0 + r) * DIMC + c0 + 4 * q);
            float4 kv = *(const float4*)(Kb + (size_t)(n0 + r) * DIMC + d0 + 4 * q);
            lq[(4*q+0)*LPITCH + r] = (bf16_t)qv.x;
            lq[(4*q+1)*LPITCH + r] = (bf16_t)qv.y;
            lq[(4*q+2)*LPITCH + r] = (bf16_t)qv.z;
            lq[(4*q+3)*LPITCH + r] = (bf16_t)qv.w;
            lk[(4*q+0)*LPITCH + r] = (bf16_t)kv.x;
            lk[(4*q+1)*LPITCH + r] = (bf16_t)kv.y;
            lk[(4*q+2)*LPITCH + r] = (bf16_t)kv.z;
            lk[(4*q+3)*LPITCH + r] = (bf16_t)kv.w;
        }
        __syncthreads();

        v16bf afr[2], bfr[4];
        #pragma unroll
        for (int mi = 0; mi < 2; ++mi) {
            int m = wm * 32 + mi * 16 + lr;                 // ISA 16-bit A layout
            afr[mi] = load_frag(lq + m * LPITCH, 8 * lh, 16 + 8 * lh);
        }
        #pragma unroll
        for (int ni = 0; ni < 4; ++ni) {
            int n = wn * 64 + ni * 16 + lr;                 // ISA 16-bit B layout
            bfr[ni] = load_frag(lk + n * LPITCH, 8 * lh, 16 + 8 * lh);
        }
        #pragma unroll
        for (int mi = 0; mi < 2; ++mi)
            #pragma unroll
            for (int ni = 0; ni < 4; ++ni)
                acc[mi][ni] = __builtin_amdgcn_wmma_f32_16x16x32_bf16(
                    false, afr[mi], false, bfr[ni], (short)0, acc[mi][ni], false, false);
    }

    float* Sb = S + (size_t)b * DIMC * DIMC;
    #pragma unroll
    for (int mi = 0; mi < 2; ++mi)
        #pragma unroll
        for (int ni = 0; ni < 4; ++ni) {
            int dcol  = d0 + wn * 64 + ni * 16 + lr;
            int mbase = c0 + wm * 32 + mi * 16 + 8 * lh;
            #pragma unroll
            for (int vr = 0; vr < 8; ++vr)
                Sb[(size_t)(mbase + vr) * DIMC + dcol] = acc[mi][ni][vr];
        }
}

// ---------------------------------------------------------------------------
// Kernel 2: in-place row softmax of S * SCALE (one 768-wide row per block)
// ---------------------------------------------------------------------------
__global__ __launch_bounds__(256)
void softmax_kernel(float* __restrict__ S)
{
    __shared__ float red[256];
    float* p = S + (size_t)blockIdx.x * DIMC;
    const int t = threadIdx.x;

    float x0 = p[t]       * SCALEF;
    float x1 = p[t + 256] * SCALEF;
    float x2 = p[t + 512] * SCALEF;

    red[t] = fmaxf(x0, fmaxf(x1, x2));
    __syncthreads();
    for (int s = 128; s > 0; s >>= 1) {
        if (t < s) red[t] = fmaxf(red[t], red[t + s]);
        __syncthreads();
    }
    float m = red[0];
    __syncthreads();

    float e0 = __expf(x0 - m), e1 = __expf(x1 - m), e2 = __expf(x2 - m);
    red[t] = e0 + e1 + e2;
    __syncthreads();
    for (int s = 128; s > 0; s >>= 1) {
        if (t < s) red[t] += red[t + s];
        __syncthreads();
    }
    float inv = 1.0f / red[0];
    p[t]       = e0 * inv;
    p[t + 256] = e1 * inv;
    p[t + 512] = e2 * inv;
}

// ---------------------------------------------------------------------------
// Kernel 3:  O[b,c,n] = sum_d P[b,c,d] * V[b,n,d]
// A = P fp32 (cvt-staged by all waves).  B = V bf16: staged by the Tensor
// Data Mover — wave 0 issues one tensor_load_to_lds per K-step; the D#'s
// pad_enable (16 dwords data + 4 dwords pad) reproduces the 80 B LDS pitch.
// ---------------------------------------------------------------------------
__global__ __launch_bounds__(256)
void pv_gemm_kernel(const float* __restrict__ P, const bf16_t* __restrict__ Vb,
                    float* __restrict__ O)
{
    __shared__ __attribute__((aligned(16))) bf16_t la[TILE * LPITCH];
    __shared__ __attribute__((aligned(16))) bf16_t lb[TILE * LPITCH];

    const int b    = blockIdx.z;
    const int c0   = blockIdx.x * TILE;
    const int n0   = blockIdx.y * TILE;
    const int t    = threadIdx.x;
    const int wave = t >> 5, lane = t & 31;
    const int wm   = wave & 3, wn = wave >> 2;
    const int lr   = lane & 15, lh = lane >> 4;

    const float*  Pb  = P  + (size_t)b * DIMC * DIMC;
    const bf16_t* Vbb = Vb + (size_t)b * SEQN * DIMC;

    v8f acc[2][4] = {};

    for (int d0 = 0; d0 < DIMC; d0 += KT) {
        if (d0 + KT < DIMC) {
            __builtin_prefetch(Pb  + (size_t)c0 * DIMC + d0 + KT, 0, 1);
            __builtin_prefetch(Vbb + (size_t)n0 * DIMC + d0 + KT, 0, 1);
        }
        __syncthreads();

#if HAVE_TDM
        if (wave == 0) {
            // Tensor DMA: V[n0..n0+127][d0..d0+31] bf16 -> lb with padded pitch.
            unsigned lds_base = (unsigned)(uintptr_t)&lb[0];
            unsigned long long ga =
                (unsigned long long)(uintptr_t)(Vbb + (size_t)n0 * DIMC + d0);
            u32x4 g0 = { 1u,                                   // count=1 (valid D#)
                         lds_base,                             // lds_addr
                         (unsigned)ga,                         // global_addr[31:0]
                         (unsigned)((ga >> 32) & 0x01FFFFFFull) | 0x80000000u }; // [56:32] | type=2
            i32x8 g1 = { (int)((1u << 16)      // data_size = 2 bytes
                             | (1u << 20)      // pad_enable
                             | (3u << 22)      // pad_interval: 16 dwords (64 B)
                             | (3u << 25)),    // pad_amount:    4 dwords (16 B)
                         (int)((unsigned)DIMC << 16),          // tensor_dim0 = 768
                         (int)((unsigned)SEQN << 16),          // tensor_dim1 = 4096
                         (int)((unsigned)KT << 16),            // tile_dim0 = 32
                         TILE,                                  // tile_dim1 = 128
                         DIMC,                                  // tensor_dim0_stride = 768
                         0, 0 };
            i32x4 gz = { 0, 0, 0, 0 };
#if __clang_major__ >= 23
            i32x8 gz8 = { 0, 0, 0, 0, 0, 0, 0, 0 };
            __builtin_amdgcn_tensor_load_to_lds(g0, g1, gz, gz, gz8, 0);
#else
            __builtin_amdgcn_tensor_load_to_lds(g0, g1, gz, gz, 0);
#endif
        }
#endif
        // All waves stage the A (P) tile while the TDM streams V.
        #pragma unroll
        for (int i = 0; i < 4; ++i) {
            int idx = t + i * 256;          // 0..1023
            int r   = idx >> 3;             // tile row 0..127
            int q   = idx & 7;              // group of 4 d's
            float4 av = *(const float4*)(Pb + (size_t)(c0 + r) * DIMC + d0 + 4 * q);
            v4bf ap = { (bf16_t)av.x, (bf16_t)av.y, (bf16_t)av.z, (bf16_t)av.w };
            *(v4bf*)&la[r * LPITCH + 4 * q] = ap;
        }
#if HAVE_TDM
        if (wave == 0) __builtin_amdgcn_s_wait_tensorcnt(0);
#else
        #pragma unroll
        for (int i = 0; i < 2; ++i) {
            int idx = t + i * 256;          // 0..511
            int r   = idx >> 2;             // row 0..127
            int ch  = idx & 3;              // 16 B chunk within 64 B row
            *(v8bf*)&lb[r * LPITCH + ch * 8] =
                *(const v8bf*)(Vbb + (size_t)(n0 + r) * DIMC + d0 + ch * 8);
        }
#endif
        __syncthreads();

        v16bf afr[2], bfr[4];
        #pragma unroll
        for (int mi = 0; mi < 2; ++mi) {
            int m = wm * 32 + mi * 16 + lr;
            afr[mi] = load_frag(la + m * LPITCH, 8 * lh, 16 + 8 * lh);
        }
        #pragma unroll
        for (int ni = 0; ni < 4; ++ni) {
            int n = wn * 64 + ni * 16 + lr;
            bfr[ni] = load_frag(lb + n * LPITCH, 8 * lh, 16 + 8 * lh);
        }
        #pragma unroll
        for (int mi = 0; mi < 2; ++mi)
            #pragma unroll
            for (int ni = 0; ni < 4; ++ni)
                acc[mi][ni] = __builtin_amdgcn_wmma_f32_16x16x32_bf16(
                    false, afr[mi], false, bfr[ni], (short)0, acc[mi][ni], false, false);
    }

    float* Ob = O + (size_t)b * DIMC * SEQN;
    #pragma unroll
    for (int mi = 0; mi < 2; ++mi)
        #pragma unroll
        for (int ni = 0; ni < 4; ++ni) {
            int ncol  = n0 + wn * 64 + ni * 16 + lr;
            int mbase = c0 + wm * 32 + mi * 16 + 8 * lh;
            #pragma unroll
            for (int vr = 0; vr < 8; ++vr)
                Ob[(size_t)(mbase + vr) * SEQN + ncol] = acc[mi][ni][vr];
        }
}

// ---------------------------------------------------------------------------
extern "C" void kernel_launch(void* const* d_in, const int* in_sizes, int n_in,
                              void* d_out, int out_size, void* d_ws, size_t ws_size,
                              hipStream_t stream)
{
    (void)in_sizes; (void)n_in; (void)out_size; (void)ws_size;
    const float* x1 = (const float*)d_in[0];   // Q
    const float* x2 = (const float*)d_in[1];   // K = V
    float* out = (float*)d_out;

    // workspace layout: [0, 18.9MB) S fp32 logits/probs ; then bf16 copy of V
    float*  S  = (float*)d_ws;
    bf16_t* Vb = (bf16_t*)((char*)d_ws + (size_t)NBATCH * DIMC * DIMC * sizeof(float));

    dim3 blk(256);
    cvt_bf16_kernel<<<dim3((NBATCH * SEQN * DIMC) / (4 * 256)), blk, 0, stream>>>(x2, Vb);
    qk_gemm_kernel<<<dim3(DIMC / TILE, DIMC / TILE, NBATCH), blk, 0, stream>>>(x1, x2, S);
    softmax_kernel<<<dim3(NBATCH * DIMC), blk, 0, stream>>>(S);
    pv_gemm_kernel<<<dim3(DIMC / TILE, SEQN / TILE, NBATCH), blk, 0, stream>>>(S, Vb, out);
}
